// LinearAttention_13503377178736
// MI455X (gfx1250) — matrix operand order repaired and checked by reading, not compiled, not run on hardware
//
#include <hip/hip_runtime.h>

#define SEQL   2048
#define BATCHN 4
#define NHEADS 16
#define HD     64
#define FEATN  16
#define CEXP   273
#define FPAD   288          // logical padded feature dim (multiple of 32)
#define QFS    296          // LDS row stride for Qf/Kf (bank-conflict pad, 16B aligned)
#define CH     64           // chunk length
#define DM     1024
#define QKVN   1536         // 256 q + 256 k + 1024 v
#define MROWS  (BATCHN * SEQL)   // 8192

typedef __attribute__((ext_vector_type(16))) __bf16 v16bf;
typedef __attribute__((ext_vector_type(8)))  __bf16 v8bf;
typedef __attribute__((ext_vector_type(8)))  float  v8f;
typedef __attribute__((vector_size(16)))     int    v4i;

#if defined(__gfx1250__) && __has_builtin(__builtin_amdgcn_global_load_async_to_lds_b128) && __has_builtin(__builtin_amdgcn_s_wait_asynccnt)
#define USE_ASYNC 1
#else
#define USE_ASYNC 0
#endif

#if USE_ASYNC
typedef __attribute__((address_space(1))) v4i* gptr_v4i;
typedef __attribute__((address_space(3))) v4i* lptr_v4i;
__device__ __forceinline__ void async_b128(const void* g, void* l) {
  __builtin_amdgcn_global_load_async_to_lds_b128(
      (gptr_v4i)const_cast<void*>(g), (lptr_v4i)l, 0, 0);
}
__device__ __forceinline__ void async_wait0() {
  __builtin_amdgcn_s_wait_asynccnt(0);
}
#endif

__device__ __forceinline__ v8f wmma_bf16(v16bf a, v16bf b, v8f c) {
  return __builtin_amdgcn_wmma_f32_16x16x32_bf16(false, a, false, b, (short)0, c,
                                                 false, false);
}

// A fragment (16x32 bf16), source row-major with given element stride.
// lane L: row=L&15, K in {kh..kh+15} with kh = (L>=16)*16.
__device__ __forceinline__ v16bf load_a_frag(const __bf16* base, int stride, int lane) {
  const __bf16* rp = base + (lane & 15) * stride + ((lane >> 4) << 4);
  v16bf a;
#pragma unroll
  for (int i = 0; i < 16; ++i) a[i] = rp[i];
  return a;
}

// B fragment (32x16 bf16) from K-contiguous ("transposed") storage:
// element (k, n) lives at base[n*stride + k]. lane L: col=L&15, kbase=(L>>4)*16.
__device__ __forceinline__ v16bf load_b_frag(const __bf16* base, int stride, int lane) {
  const __bf16* cp = base + (lane & 15) * stride + ((lane >> 4) << 4);
  v16bf b;
#pragma unroll
  for (int i = 0; i < 16; ++i) b[i] = cp[i];
  return b;
}

// ---------------- convert / pack ----------------

__global__ void cvt_f32_bf16(const float* __restrict__ in, __bf16* __restrict__ out, int n) {
  int stride = gridDim.x * blockDim.x;
  for (int i = blockIdx.x * blockDim.x + threadIdx.x; i < n; i += stride)
    out[i] = (__bf16)in[i];
}

__global__ void pack_wqkv(const float* __restrict__ Wq, const float* __restrict__ Wk,
                          const float* __restrict__ Wv, const float* __restrict__ bq,
                          const float* __restrict__ bk, const float* __restrict__ bv,
                          __bf16* __restrict__ Wcat, float* __restrict__ bcat) {
  int stride = gridDim.x * blockDim.x;
  for (int i = blockIdx.x * blockDim.x + threadIdx.x; i < DM * QKVN; i += stride) {
    int k = i / QKVN, c = i - k * QKVN;
    float v = (c < 256) ? Wq[k * 256 + c]
            : (c < 512) ? Wk[k * 256 + (c - 256)]
                        : Wv[k * 1024 + (c - 512)];
    Wcat[i] = (__bf16)v;
  }
  for (int c = blockIdx.x * blockDim.x + threadIdx.x; c < QKVN; c += stride)
    bcat[c] = (c < 256) ? bq[c] : (c < 512) ? bk[c - 256] : bv[c - 512];
}

// ---------------- bf16 GEMM: C[M,N] = A[M,K] @ B[K,N] + bias ----------------
// 64x64 block tile, ping-pong LDS double buffering, async A-tile staging.

__global__ __launch_bounds__(256) void gemm_bf16f32(
    const __bf16* __restrict__ A, const __bf16* __restrict__ B,
    const float* __restrict__ bias, float* __restrict__ C,
    int M, int N, int K) {
  __shared__ __bf16 As[2][64][40];   // stride 40: conflict-free, rows 16B aligned
  __shared__ __bf16 Bs[2][64][40];   // transposed: [n][k]
  int tid = threadIdx.x, lane = tid & 31, w = tid >> 5;
  int wm = w & 3, wn = w >> 2;
  int m0 = blockIdx.y << 6, n0 = blockIdx.x << 6;
  int ar = tid >> 2, ac = (tid & 3) << 3;      // A: 64 rows x 4 loaders x 8 elems
  int bk = tid >> 3, bn = (tid & 7) << 3;      // B: 32 rows x 8 loaders x 8 elems
  const __bf16* ag = A + (size_t)(m0 + ar) * K + ac;
  const __bf16* bg = B + (size_t)bk * N + n0 + bn;
  const int ksteps = K >> 5;
  v8f c0 = {}, c1 = {};

  // prologue: stage tile 0
#if USE_ASYNC
  async_b128(ag, &As[0][ar][ac]);
#else
  *(v8bf*)&As[0][ar][ac] = *(const v8bf*)ag;
#endif
  {
    v8bf bv = *(const v8bf*)bg;
#pragma unroll
    for (int j = 0; j < 8; ++j) Bs[0][bn + j][bk] = bv[j];
  }
#if USE_ASYNC
  async_wait0();
#endif
  __syncthreads();

  for (int kt = 0; kt < ksteps; ++kt) {
    int cur = kt & 1, nxt = cur ^ 1;
    bool more = (kt + 1) < ksteps;
    v8bf bv;
    if (more) {
#if USE_ASYNC
      async_b128(ag + (kt + 1) * 32, &As[nxt][ar][ac]);
#else
      *(v8bf*)&As[nxt][ar][ac] = *(const v8bf*)(ag + (kt + 1) * 32);
#endif
      bv = *(const v8bf*)(bg + (size_t)((kt + 1) * 32) * N);
      __builtin_prefetch(bg + (size_t)((kt + 2) * 32) * N, 0, 1);  // global_prefetch_b8
    }
    // compute current tile while next tile is in flight
    v16bf af = load_a_frag(&As[cur][wm << 4][0], 40, lane);
    v16bf b0 = load_b_frag(&Bs[cur][wn << 5][0], 40, lane);
    v16bf b1 = load_b_frag(&Bs[cur][(wn << 5) + 16][0], 40, lane);
    c0 = wmma_bf16(af, b0, c0);
    c1 = wmma_bf16(af, b1, c1);
    if (more) {
#pragma unroll
      for (int j = 0; j < 8; ++j) Bs[nxt][bn + j][bk] = bv[j];
#if USE_ASYNC
      async_wait0();
#endif
    }
    __syncthreads();
  }

  int col = n0 + (wn << 5) + (lane & 15);
  int row = m0 + (wm << 4) + ((lane >> 4) << 3);
  float bb0 = bias[col];
  float bb1 = bias[col + 16];
#pragma unroll
  for (int r = 0; r < 8; ++r) {
    C[(size_t)(row + r) * N + col]      = c0[r] + bb0;
    C[(size_t)(row + r) * N + col + 16] = c1[r] + bb1;
  }
}

// ---------------- chunkwise linear-attention scan ----------------
// One block per (batch, head). fp32 KV state resident in LDS for all 32 chunks.

#define SCAN_LDS_BYTES 175488

__global__ __launch_bounds__(256) void scan_chunked(
    const float* __restrict__ QKV,   // [MROWS][QKVN] fp32
    const float* __restrict__ kv0,   // [B*H][CEXP][HD]
    const float* __restrict__ kst0,  // [B*H][CEXP]
    __bf16* __restrict__ Obf,        // [MROWS][DM]  (attention out, pre out-proj)
    float* __restrict__ kv_out, float* __restrict__ kst_out) {
  extern __shared__ char smemRaw[];
  float*  KV  = (float*)smemRaw;            // [FPAD][HD]       73728 B
  float*  kst = KV + FPAD * HD;             // [FPAD]            1152 B
  __bf16* Qf  = (__bf16*)(kst + FPAD);      // [CH][QFS]        37888 B
  __bf16* Kf  = Qf + CH * QFS;              // [CH][QFS]        37888 B
  __bf16* Vt  = Kf + CH * QFS;              // [HD][CH] (Vt[c][t]) 8192 B
  __bf16* Ab  = Vt + HD * CH;               // [CH][CH] masked A   8192 B
  float*  den = (float*)(Ab + CH * CH);     // [CH]               256 B
  float*  qs  = den + CH;                   // [CH][FEATN]       4096 B (also den scratch)
  float*  ks  = qs + CH * FEATN;            // [CH][FEATN]       4096 B

  int bh = blockIdx.x;
  int b = bh >> 4, h = bh & 15;
  int tid = threadIdx.x, lane = tid & 31, w = tid >> 5;
  int wm = w & 3, wn = w >> 2;

  for (int i = tid; i < FPAD * HD; i += 256) {
    int f = i / HD, c = i - f * HD;
    KV[i] = (f < CEXP) ? kv0[((size_t)bh * CEXP + f) * HD + c] : 0.f;
  }
  for (int f = tid; f < FPAD; f += 256)
    kst[f] = (f < CEXP) ? kst0[(size_t)bh * CEXP + f] : 0.f;
  __syncthreads();

  const float s1 = 0.5f;            // 1/d^(1/4), d=16
  const float s2 = 0.17677669529f;  // 1/(sqrt(2)*sqrt(16))

  for (int ch = 0; ch < SEQL / CH; ++ch) {
    size_t row0 = (size_t)b * SEQL + (size_t)ch * CH;
    // stage q16/k16 (fp32): 4 floats (16B) per thread, async direct to LDS
#if USE_ASYNC
    {
      int t = tid >> 2, j4 = (tid & 3) << 2;
      const float* rp = QKV + (row0 + t) * QKVN + h * FEATN + j4;
      async_b128(rp, qs + t * FEATN + j4);
      async_b128(rp + 256, ks + t * FEATN + j4);
    }
#else
    for (int i = tid; i < CH * FEATN; i += 256) {
      int t = i >> 4, j = i & 15;
      const float* rp = QKV + (row0 + t) * QKVN;
      qs[i] = rp[h * FEATN + j];
      ks[i] = rp[256 + h * FEATN + j];
    }
#endif
    // V transposed (bf16) via registers
    for (int i = tid; i < CH * HD; i += 256) {
      int t = i >> 6, c = i & 63;
      Vt[c * CH + t] = (__bf16)QKV[(row0 + t) * QKVN + 512 + h * HD + c];
    }
#if USE_ASYNC
    async_wait0();
#endif
    __syncthreads();
    // Taylor feature map -> bf16 Qf/Kf, rows padded with zeros to FPAD
    for (int i = tid; i < CH * FPAD; i += 256) {
      int t = i / FPAD, f = i - t * FPAD;
      float qv, kv_;
      if (f == 0)          { qv = 1.f; kv_ = 1.f; }
      else if (f <= FEATN) { qv = qs[t * FEATN + f - 1] * s1;
                             kv_ = ks[t * FEATN + f - 1] * s1; }
      else if (f < CEXP)   { int ff = f - 1 - FEATN; int ii = ff >> 4, jj = ff & 15;
                             qv  = qs[t * FEATN + ii] * qs[t * FEATN + jj] * s2;
                             kv_ = ks[t * FEATN + ii] * ks[t * FEATN + jj] * s2; }
      else                 { qv = 0.f; kv_ = 0.f; }
      Qf[t * QFS + f] = (__bf16)qv;
      Kf[t * QFS + f] = (__bf16)kv_;
    }
    __syncthreads();
    // A = tril(Qf @ Kf^T): 8 waves cover 64x64 (2 tiles each)
    {
      v8f a0 = {}, a1 = {};
      for (int kk = 0; kk < FPAD; kk += 32) {
        v16bf af = load_a_frag(Qf + (wm << 4) * QFS + kk, QFS, lane);
        v16bf b0 = load_b_frag(Kf + (wn << 5) * QFS + kk, QFS, lane);
        v16bf b1 = load_b_frag(Kf + ((wn << 5) + 16) * QFS + kk, QFS, lane);
        a0 = wmma_bf16(af, b0, a0);
        a1 = wmma_bf16(af, b1, a1);
      }
      int c0i = (wn << 5) + (lane & 15);
      int r0i = (wm << 4) + ((lane >> 4) << 3);
#pragma unroll
      for (int r = 0; r < 8; ++r) {
        int rr = r0i + r;
        Ab[rr * CH + c0i]      = (__bf16)((c0i <= rr) ? a0[r] : 0.f);
        Ab[rr * CH + c0i + 16] = (__bf16)((c0i + 16 <= rr) ? a1[r] : 0.f);
      }
    }
    __syncthreads();
    // den[t] = rowsum(A_masked) + Qf[t].kst + 1e-6; 4 partials per row (qs is dead)
    {
      int row = tid >> 2, part = tid & 3;
      float s = 0.f;
      int a0i = part << 4;
      for (int t2 = a0i; t2 < a0i + 16; ++t2) s += (float)Ab[row * CH + t2];
      int f0i = part * 72;                       // 288/4
      for (int f = f0i; f < f0i + 72; ++f) s += (float)Qf[row * QFS + f] * kst[f];
      qs[tid] = s;
    }
    __syncthreads();
    if (tid < CH)
      den[tid] = qs[tid * 4] + qs[tid * 4 + 1] + qs[tid * 4 + 2] + qs[tid * 4 + 3] + 1e-6f;
    __syncthreads();
    // O = A@V + Qf@KV, normalize, store bf16 into attention-out buffer
    {
      v8f o0 = {}, o1 = {};
      for (int kk = 0; kk < CH; kk += 32) {
        v16bf af = load_a_frag(Ab + (wm << 4) * CH + kk, CH, lane);
        v16bf b0 = load_b_frag(Vt + (wn << 5) * CH + kk, CH, lane);
        v16bf b1 = load_b_frag(Vt + ((wn << 5) + 16) * CH + kk, CH, lane);
        o0 = wmma_bf16(af, b0, o0);
        o1 = wmma_bf16(af, b1, o1);
      }
      for (int kk = 0; kk < FPAD; kk += 32) {
        v16bf af = load_a_frag(Qf + (wm << 4) * QFS + kk, QFS, lane);
        int kb = kk + ((lane >> 4) << 4);
        int c0i = (wn << 5) + (lane & 15);
        v16bf b0, b1;
#pragma unroll
        for (int i = 0; i < 16; ++i) {        // fp32 state -> bf16 B fragment
          b0[i] = (__bf16)KV[(kb + i) * HD + c0i];
          b1[i] = (__bf16)KV[(kb + i) * HD + c0i + 16];
        }
        o0 = wmma_bf16(af, b0, o0);
        o1 = wmma_bf16(af, b1, o1);
      }
      int c0i = (wn << 5) + (lane & 15);
      int r0i = (wm << 4) + ((lane >> 4) << 3);
#pragma unroll
      for (int r = 0; r < 8; ++r) {
        int rr = r0i + r;
        float dinv = 1.f / den[rr];
        size_t gr = (row0 + rr) * (size_t)DM + (size_t)h * HD;
        Obf[gr + c0i]      = (__bf16)(o0[r] * dinv);
        Obf[gr + c0i + 16] = (__bf16)(o1[r] * dinv);
      }
    }
    __syncthreads();
    // KV += Kf^T @ V : 18x4 tiles of 16x16, 9 per wave, C operand from LDS
    for (int tile = w; tile < 72; tile += 8) {
      int f0 = (tile >> 2) << 4, c0 = (tile & 3) << 4;
      int col = c0 + (lane & 15);
      int rb  = f0 + ((lane >> 4) << 3);
      v8f cc;
#pragma unroll
      for (int r = 0; r < 8; ++r) cc[r] = KV[(rb + r) * HD + col];
      int kh = (lane >> 4) << 4;
      int fr = f0 + (lane & 15);
      for (int kk = 0; kk < CH; kk += 32) {
        v16bf af;            // A = Kf^T: a[i] = Kf[time][feature fr]
#pragma unroll
        for (int i = 0; i < 8; ++i) {
          af[i]     = Kf[(kk + kh + i) * QFS + fr];
          af[i + 8] = Kf[(kk + kh + 8 + i) * QFS + fr];
        }
        v16bf b0 = load_b_frag(Vt + c0 * CH + kk, CH, lane);
        cc = wmma_bf16(af, b0, cc);
      }
#pragma unroll
      for (int r = 0; r < 8; ++r) KV[(rb + r) * HD + col] = cc[r];
    }
    // kst += column sums of Kf
    for (int f = tid; f < FPAD; f += 256) {
      float s = 0.f;
      for (int t = 0; t < CH; ++t) s += (float)Kf[t * QFS + f];
      kst[f] += s;
    }
    __syncthreads();
  }
  // final states
  for (int i = tid; i < CEXP * HD; i += 256) {
    int f = i / HD, c = i - f * HD;
    kv_out[((size_t)bh * CEXP + f) * HD + c] = KV[f * HD + c];
  }
  for (int f = tid; f < CEXP; f += 256)
    kst_out[(size_t)bh * CEXP + f] = kst[f];
}

// ---------------- launch ----------------

extern "C" void kernel_launch(void* const* d_in, const int* in_sizes, int n_in,
                              void* d_out, int out_size, void* d_ws, size_t ws_size,
                              hipStream_t stream) {
  const float* x   = (const float*)d_in[0];
  const float* kv0 = (const float*)d_in[1];
  const float* k0  = (const float*)d_in[2];
  const float* Wq  = (const float*)d_in[3];
  const float* bq  = (const float*)d_in[4];
  const float* Wk  = (const float*)d_in[5];
  const float* bk  = (const float*)d_in[6];
  const float* Wv  = (const float*)d_in[7];
  const float* bv  = (const float*)d_in[8];
  const float* Wo  = (const float*)d_in[9];
  const float* bo  = (const float*)d_in[10];

  float* y       = (float*)d_out;
  float* kv_out  = y + (size_t)MROWS * DM;
  float* kst_out = kv_out + (size_t)BATCHN * NHEADS * CEXP * HD;

  char* ws = (char*)d_ws;
  __bf16* xb   = (__bf16*)ws; ws += (size_t)MROWS * DM * 2;
  __bf16* Wcat = (__bf16*)ws; ws += (size_t)DM * QKVN * 2;
  __bf16* Wob  = (__bf16*)ws; ws += (size_t)DM * DM * 2;
  float*  bcat = (float*)ws;  ws += (size_t)QKVN * 4;
  float*  QKV  = (float*)ws;  ws += (size_t)MROWS * QKVN * 4;
  __bf16* Obf  = (__bf16*)ws; ws += (size_t)MROWS * DM * 2;

  cvt_f32_bf16<<<2048, 256, 0, stream>>>(x, xb, MROWS * DM);
  cvt_f32_bf16<<<1024, 256, 0, stream>>>(Wo, Wob, DM * DM);
  pack_wqkv<<<1024, 256, 0, stream>>>(Wq, Wk, Wv, bq, bk, bv, Wcat, bcat);

  dim3 g1(QKVN / 64, MROWS / 64);
  gemm_bf16f32<<<g1, 256, 0, stream>>>(xb, Wcat, bcat, QKV, MROWS, QKVN, DM);

  (void)hipFuncSetAttribute((const void*)scan_chunked,
                            hipFuncAttributeMaxDynamicSharedMemorySize,
                            SCAN_LDS_BYTES);
  scan_chunked<<<BATCHN * NHEADS, 256, SCAN_LDS_BYTES, stream>>>(
      QKV, kv0, k0, Obf, kv_out, kst_out);

  dim3 g2(DM / 64, MROWS / 64);
  gemm_bf16f32<<<g2, 256, 0, stream>>>(Obf, Wob, bo, y, MROWS, DM, DM);
}
